// MultiHeadAttention_24807731102268
// MI455X (gfx1250) — matrix operand order repaired and checked by reading, not compiled, not run on hardware
//
#include <hip/hip_runtime.h>
#include <hip/hip_fp16.h>

// Problem constants (match reference)
#define BATCH  4
#define SEQ    2048
#define DMODEL 1024
#define NHEAD  16
#define HDIM   64

typedef __attribute__((ext_vector_type(16))) _Float16 v16h;
typedef __attribute__((ext_vector_type(8)))  _Float16 v8h;
typedef __attribute__((ext_vector_type(4)))  _Float16 v4h;
typedef __attribute__((ext_vector_type(8)))  float    v8f;
typedef __attribute__((ext_vector_type(4)))  unsigned int v4u;
typedef __attribute__((ext_vector_type(4)))  int      v4i;
typedef __attribute__((ext_vector_type(8)))  int      v8i;

#if defined(__has_builtin)
#  if __has_builtin(__builtin_amdgcn_permlane16)
#    define USE_PERMLANE16 1
#  endif
#  if __has_builtin(__builtin_amdgcn_tensor_load_to_lds) && \
      __has_builtin(__builtin_amdgcn_s_wait_tensorcnt)
#    define USE_TDM 1
#  endif
#endif

// ---- CDNA5 async global->LDS copy (ASYNCcnt path) --------------------------
// LDS byte address = low 32 bits of the flat pointer (aperture truncation rule).
static __device__ __forceinline__ void async_copy_b128(void* lds, const void* gsrc) {
  unsigned l = (unsigned)(unsigned long long)lds;
  asm volatile("global_load_async_to_lds_b128 %0, %1, off"
               :: "v"(l), "v"(gsrc) : "memory");
}
// Wait until ASYNCcnt <= N (N=0: all done; N=k: drain all but the newest k).
template <int N>
static __device__ __forceinline__ void wait_async_le() {
  asm volatile("s_wait_asynccnt %0" :: "i"(N) : "memory");
}

#ifdef USE_TDM
// ---- Tensor Data Mover: one instruction moves a 64x64 f16 tile to LDS ------
// D# per CDNA5 ISA ch.8: 2-D tensor, data_size=2B, LDS pad 4 DWORDs after every
// 32 DWORDs (reproduces the 72-element LDS row pitch used by the fragments).
static __device__ __forceinline__ void tdm_load_tile64x64_f16(
    void* lds, const void* gsrc, unsigned rowStrideElems) {
  const unsigned long long ga = (unsigned long long)gsrc;
  v4u g0;
  g0[0] = 1u;                                        // count=1, user descriptor
  g0[1] = (unsigned)(unsigned long long)lds;         // lds_addr (bytes)
  g0[2] = (unsigned)ga;                              // global_addr[31:0]
  g0[3] = (unsigned)((ga >> 32) & 0x01FFFFFFull)     // global_addr[56:32]
          | (2u << 30);                              // type = 2 ("image")
  v8i g1;
  g1[0] = (1 << 16)                                  // data_size = 2 bytes
        | (1 << 20)                                  // pad_enable
        | (4 << 22)                                  // pad_interval: 32 DWORDs
        | (3 << 25);                                 // pad_amount: 4 DWORDs
  g1[1] = 64 << 16;                                  // tensor_dim0 = 64
  g1[2] = 64 << 16;                                  // tensor_dim1 = 64 (lo)
  g1[3] = 64 << 16;                                  // tile_dim0 = 64
  g1[4] = 64;                                        // tile_dim1 = 64, tile_dim2 = 0
  g1[5] = (int)rowStrideElems;                       // tensor_dim0_stride[31:0]
  g1[6] = 0;
  g1[7] = 0;
  const v4i z4 = {0, 0, 0, 0};
  const v8i z8 = {0, 0, 0, 0, 0, 0, 0, 0};
  __builtin_amdgcn_tensor_load_to_lds(g0, g1, z4, z4, z8, 0);
}
#endif

static __device__ __forceinline__ v8f zero8() {
  v8f z;
#pragma unroll
  for (int i = 0; i < 8; ++i) z[i] = 0.0f;
  return z;
}

static __device__ __forceinline__ v8f wmma32(v16h a, v16h b, v8f c) {
  // D = A(16x32 f16) * B(32x16 f16) + C(16x16 f32)
  return __builtin_amdgcn_wmma_f32_16x16x32_f16(false, a, false, b,
                                                (short)0, c, false, false);
}

// A-fragment (16x32 f16): lanes 0-15 K={0..7,16..23}, lanes 16-31 K={8..15,24..31}
static __device__ __forceinline__ v16h ld_afrag(const _Float16* rowbase, int half) {
  v8h lo = *(const v8h*)(rowbase + half * 8);
  v8h hi = *(const v8h*)(rowbase + half * 8 + 16);
  return __builtin_shufflevector(lo, hi, 0, 1, 2, 3, 4, 5, 6, 7,
                                 8, 9, 10, 11, 12, 13, 14, 15);
}

// B-fragment (32x16 f16) from K-contiguous transposed tile Bt[n][k]:
// lanes 0-15 K={0..7,16..23}, lanes 16-31 K={8..15,24..31} (same as A)
static __device__ __forceinline__ v16h ld_bfrag(const _Float16* rowbase, int half) {
  v8h lo = *(const v8h*)(rowbase + half * 8);
  v8h hi = *(const v8h*)(rowbase + half * 8 + 16);
  return __builtin_shufflevector(lo, hi, 0, 1, 2, 3, 4, 5, 6, 7,
                                 8, 9, 10, 11, 12, 13, 14, 15);
}

// ---- lane-xor within 16-lane rows: v_permlane16_b32 (VALU) or shuffle ------
static constexpr unsigned xorsel(int m, int base) {
  unsigned v = 0;
  for (int j = 0; j < 8; ++j)
    v |= ((unsigned)(((base + j) ^ m) & 0xF)) << (4 * j);
  return v;
}
template <int MASK>
static __device__ __forceinline__ float xor16f(float x) {
#ifdef USE_PERMLANE16
  const unsigned u = __float_as_uint(x);
  const unsigned r = __builtin_amdgcn_permlane16(u, u, xorsel(MASK, 0),
                                                 xorsel(MASK, 8), false, false);
  return __uint_as_float(r);
#else
  return __shfl_xor(x, MASK, 16);
#endif
}

// ---------------------------------------------------------------------------
// One-pass fp32 -> f16 conversion (bandwidth bound, done once per tensor)
// ---------------------------------------------------------------------------
__global__ __launch_bounds__(256) void cvt_f32_f16(const float* __restrict__ s,
                                                   _Float16* __restrict__ d, int n4) {
  const int i = blockIdx.x * 256 + threadIdx.x;
  if (i >= n4) return;
  const float4 a = ((const float4*)s)[i];
  v4h h;
  h[0] = (_Float16)a.x; h[1] = (_Float16)a.y;
  h[2] = (_Float16)a.z; h[3] = (_Float16)a.w;
  *(v4h*)(d + (size_t)i * 4) = h;
}

// fp32 W[K][N] -> f16 Wt[N][K] (once; makes GEMM B-fragments contiguous-K)
__global__ __launch_bounds__(128) void transpose_cvt_f16(
    const float* __restrict__ W, _Float16* __restrict__ Wt, int K, int N) {
  constexpr int LDT = 72;
  __shared__ _Float16 T[64 * LDT];  // [n][k]
  const int tid = threadIdx.x;
  const int k0 = blockIdx.x * 64, n0 = blockIdx.y * 64;
  {
    const int kr = tid >> 4;          // 0..7
    const int nc = (tid & 15) * 4;    // 0..60
#pragma unroll
    for (int p = 0; p < 8; ++p) {
      const int k = kr + p * 8;
      const float4 w = *(const float4*)(W + (size_t)(k0 + k) * N + n0 + nc);
      T[(nc + 0) * LDT + k] = (_Float16)w.x;
      T[(nc + 1) * LDT + k] = (_Float16)w.y;
      T[(nc + 2) * LDT + k] = (_Float16)w.z;
      T[(nc + 3) * LDT + k] = (_Float16)w.w;
    }
  }
  __syncthreads();
  {
    const int r = tid >> 1, c0 = (tid & 1) * 32;
    uint4* dst = (uint4*)(Wt + (size_t)(n0 + r) * K + k0 + c0);
    const uint4* src = (const uint4*)(&T[r * LDT + c0]);
#pragma unroll
    for (int i = 0; i < 4; ++i) dst[i] = src[i];
  }
}

// ---------------------------------------------------------------------------
// 128x64-block GEMM: Out = A[M,K] * Bt[N,K]^T, all-f16, f32 accumulate.
// Double-buffered async global->LDS staging (copy overlaps WMMA), K-step 64.
// 4 waves; wave w computes a 32x64 tile (16 WMMAs per K-step).
// OUT_MODE: 0 = f16 [M,N]; 1 = f32 [M,N]; 2 = f16 V^T layout [b][n][s]
// Epilogue is re-laid out through LDS for coalesced 16B global stores.
// ---------------------------------------------------------------------------
template <int OUT_MODE>
__global__ __launch_bounds__(128) void gemm_wmma_f16(
    const _Float16* __restrict__ A, const _Float16* __restrict__ Bt,
    void* __restrict__ Outp, int M, int N, int K, float outScale) {
  constexpr int LDT  = 72;
  constexpr int ASZ  = 128 * LDT;         // 9216 f16 per A tile
  constexpr int BUFS = ASZ + 64 * LDT;    // 13824 f16 per (A+B) buffer
  // [A0 128x72 | B0 64x72 | A1 128x72 | B1 64x72] = 27648 f16 = 54 KiB
  __shared__ __align__(16) _Float16 smem[2 * BUFS];

  const int tid  = threadIdx.x;
  const int wid  = tid >> 5;
  const int lane = tid & 31;
  const int half = lane >> 4;
  const int l16  = lane & 15;
  const int m0 = blockIdx.x * 128;
  const int n0 = blockIdx.y * 64;
  const int rb = tid >> 1;             // B staging row 0..63
  const int cb = (tid & 1) * 32;       // B staging col base

  // 12 async b128 ops per thread per K-step (A: 8, B: 4)
  auto issue = [&](int buf, int k0) {
    _Float16* base = smem + buf * BUFS;          // runtime GEP (no const pointer table)
    const _Float16* ga = A + (size_t)(m0 + tid) * K + k0;   // 1 row (128B) per thread
    _Float16* as_ = base + tid * LDT;
#pragma unroll
    for (int i = 0; i < 8; ++i) async_copy_b128(as_ + i * 8, ga + i * 8);
    const _Float16* gb = Bt + (size_t)(n0 + rb) * K + k0 + cb;
    _Float16* bs_ = base + ASZ + rb * LDT + cb;
#pragma unroll
    for (int i = 0; i < 4; ++i) async_copy_b128(bs_ + i * 8, gb + i * 8);
  };

  v8f acc[2][4];
#pragma unroll
  for (int s = 0; s < 2; ++s)
#pragma unroll
    for (int j = 0; j < 4; ++j) acc[s][j] = zero8();

  const int steps = K / 64;
  issue(0, 0);
  for (int t = 0; t < steps; ++t) {
    const int buf = t & 1;
    if (t + 1 < steps) {
      issue(buf ^ 1, (t + 1) * 64);   // overlaps with this step's compute
      wait_async_le<12>();            // drain only the older batch
    } else {
      wait_async_le<0>();
    }
    __syncthreads();                  // tile[t] visible to all waves

    const _Float16* Ab = smem + buf * BUFS;
    const _Float16* Bb = Ab + ASZ;
    v16h af[2][2];
#pragma unroll
    for (int s = 0; s < 2; ++s)
#pragma unroll
      for (int kh = 0; kh < 2; ++kh)
        af[s][kh] = ld_afrag(&Ab[(wid * 32 + s * 16 + l16) * LDT + kh * 32], half);
#pragma unroll
    for (int j = 0; j < 4; ++j) {
#pragma unroll
      for (int kh = 0; kh < 2; ++kh) {
        const v16h bf = ld_bfrag(&Bb[(j * 16 + l16) * LDT + kh * 32], half);
        acc[0][j] = wmma32(af[0][kh], bf, acc[0][j]);
        acc[1][j] = wmma32(af[1][kh], bf, acc[1][j]);
      }
    }
    __syncthreads();                  // readers done -> buffer reusable
  }

  // ---- epilogue: re-layout via LDS, coalesced global stores ----
  if constexpr (OUT_MODE == 1) {
    float* C32 = (float*)smem;        // [128][72] f32 = 36 KiB
#pragma unroll
    for (int s = 0; s < 2; ++s)
#pragma unroll
      for (int j = 0; j < 4; ++j)
#pragma unroll
        for (int v = 0; v < 8; ++v)
          C32[(wid * 32 + s * 16 + half * 8 + v) * LDT + j * 16 + l16] =
              acc[s][j][v] * outScale;
    __syncthreads();
    float* dst = (float*)Outp + (size_t)(m0 + tid) * N + n0;
    const float4* src = (const float4*)(&C32[tid * LDT]);
#pragma unroll
    for (int i = 0; i < 16; ++i) ((float4*)dst)[i] = src[i];
  } else if constexpr (OUT_MODE == 0) {
    _Float16* C16 = smem;             // [128][72] f16
#pragma unroll
    for (int s = 0; s < 2; ++s)
#pragma unroll
      for (int j = 0; j < 4; ++j)
#pragma unroll
        for (int v = 0; v < 8; ++v)
          C16[(wid * 32 + s * 16 + half * 8 + v) * LDT + j * 16 + l16] =
              (_Float16)(acc[s][j][v] * outScale);
    __syncthreads();
    uint4* dst = (uint4*)((_Float16*)Outp + (size_t)(m0 + tid) * N + n0);
    const uint4* src = (const uint4*)(&C16[tid * LDT]);
#pragma unroll
    for (int i = 0; i < 8; ++i) dst[i] = src[i];
  } else {                            // V^T: [b][n][s], tile transposed in LDS
    constexpr int LDS2 = 136;         // 128 + 8 pad
    _Float16* C16 = smem;             // [64][136] f16
#pragma unroll
    for (int s = 0; s < 2; ++s)
#pragma unroll
      for (int j = 0; j < 4; ++j)
#pragma unroll
        for (int v = 0; v < 8; ++v)
          C16[(j * 16 + l16) * LDS2 + wid * 32 + s * 16 + half * 8 + v] =
              (_Float16)(acc[s][j][v] * outScale);
    __syncthreads();
    const int b  = m0 >> 11;          // m0 / SEQ (blocks never straddle a batch)
    const int s0 = m0 & 2047;
    const int r  = tid >> 1, c0 = (tid & 1) * 64;
    uint4* dst = (uint4*)((_Float16*)Outp +
                          (((size_t)b << 10) + n0 + r) * SEQ + s0 + c0);
    const uint4* src = (const uint4*)(&C16[r * LDS2 + c0]);
#pragma unroll
    for (int i = 0; i < 8; ++i) dst[i] = src[i];
  }
}

// ---------------------------------------------------------------------------
// Causal flash attention. Q,K f16 [B,S,H*64] (Q pre-scaled by 1/8);
// V supplied transposed f16 [B][H*64][S]. Output f16 [B,S,H*64].
// K/V tiles staged by the Tensor Data Mover (TENSORcnt, double-buffered) when
// available, else per-lane async b128 copies. 4 waves, 16 q rows each.
// ---------------------------------------------------------------------------
__global__ __launch_bounds__(128) void flash_attn_causal(
    const _Float16* __restrict__ Qw, const _Float16* __restrict__ Kw,
    const _Float16* __restrict__ Vtw, _Float16* __restrict__ Ow) {
  constexpr int LDT = 72;
  __shared__ __align__(16) _Float16 Qs[64 * LDT];      // [q][d]
  __shared__ __align__(16) _Float16 KsB[2][64 * LDT];  // [kv][d]
  __shared__ __align__(16) _Float16 VsB[2][64 * LDT];  // [dv][kv]
  __shared__ __align__(16) _Float16 Ps[64 * LDT];      // [q][kv]; reused for output

  const int tid  = threadIdx.x;
  const int wid  = tid >> 5;
  const int lane = tid & 31;
  const int half = lane >> 4;
  const int l16  = lane & 15;
  const int q0 = blockIdx.x * 64;
  const int h  = blockIdx.y;
  const int bb = blockIdx.z;
  const size_t headoff = (size_t)h * HDIM;
  const size_t rowbase = (size_t)bb * SEQ;
  const int r  = tid >> 1;            // staging row 0..63
  const int c0 = (tid & 1) * 32;

  auto issueKV = [&](int buf, int kv0) {
#ifdef USE_TDM
    if (wid == 0) {  // uniform descriptors; one TDM op per tile
      tdm_load_tile64x64_f16(&KsB[buf][0],
                             Kw + (rowbase + kv0) * DMODEL + headoff, DMODEL);
      tdm_load_tile64x64_f16(&VsB[buf][0],
                             Vtw + (((size_t)bb << 10) + headoff) * SEQ + kv0, SEQ);
    }
#else
    const _Float16* gk = Kw + (rowbase + kv0 + r) * DMODEL + headoff + c0;
    const _Float16* gv = Vtw + (((size_t)bb << 10) + headoff + r) * SEQ + kv0 + c0;
    _Float16* ks = &KsB[buf][r * LDT + c0];
    _Float16* vs = &VsB[buf][r * LDT + c0];
#pragma unroll
    for (int i = 0; i < 4; ++i) {
      async_copy_b128(ks + i * 8, gk + i * 8);
      async_copy_b128(vs + i * 8, gv + i * 8);
    }
#endif
  };

  // prologue: Q (async) then first K/V tile; overlap Q-frag loads with K/V fetch
  {
    const _Float16* g = Qw + (rowbase + q0 + r) * DMODEL + headoff + c0;
#pragma unroll
    for (int i = 0; i < 4; ++i) async_copy_b128(&Qs[r * LDT + c0 + i * 8], g + i * 8);
  }
  issueKV(0, 0);
#ifdef USE_TDM
  wait_async_le<0>();  // only Q uses ASYNCcnt here
#else
  wait_async_le<8>();  // the 4 Q ops (oldest) are complete; K/V may still fly
#endif
  // NOTE: wave w stages exactly rows [16w,16w+16) of Qs = the rows it reads,
  // so its own asynccnt suffices -- no barrier needed before the Q fragments.
  const v16h qf0 = ld_afrag(&Qs[(wid * 16 + l16) * LDT + 0], half);
  const v16h qf1 = ld_afrag(&Qs[(wid * 16 + l16) * LDT + 32], half);

  float mrun[8], lrun[8];
#pragma unroll
  for (int v = 0; v < 8; ++v) { mrun[v] = -__builtin_inff(); lrun[v] = 0.0f; }
  v8f oacc[4];
#pragma unroll
  for (int j = 0; j < 4; ++j) oacc[j] = zero8();

  const int ntiles = (q0 / 64) + 1;  // causal: kv tiles at/below the diagonal
  for (int t = 0; t < ntiles; ++t) {
    const int kv0 = t * 64;
    const int buf = t & 1;
    if (t + 1 < ntiles) {
      issueKV(buf ^ 1, (t + 1) * 64);  // overlaps with this tile's compute
#ifdef USE_TDM
      __builtin_amdgcn_s_wait_tensorcnt(2);  // drain older K/V pair (wave 0)
#else
      wait_async_le<8>();
#endif
    } else {
#ifdef USE_TDM
      __builtin_amdgcn_s_wait_tensorcnt(0);
#else
      wait_async_le<0>();
#endif
    }
    __syncthreads();                   // K/V tile[t] visible to all waves

    // S = Q*K^T (scale folded into Q): 8 WMMAs per wave
    const _Float16* Ks = KsB[buf];
    const _Float16* Vs = VsB[buf];
    v8f s4[4];
#pragma unroll
    for (int nt = 0; nt < 4; ++nt) {
      v8f sc = zero8();
      const v16h kf0 = ld_bfrag(&Ks[(nt * 16 + l16) * LDT + 0], half);
      const v16h kf1 = ld_bfrag(&Ks[(nt * 16 + l16) * LDT + 32], half);
      sc = wmma32(qf0, kf0, sc);
      sc = wmma32(qf1, kf1, sc);
      s4[nt] = sc;
    }

    // causal mask on the diagonal tile
    if (kv0 == q0) {
#pragma unroll
      for (int nt = 0; nt < 4; ++nt) {
        const int kvg = kv0 + nt * 16 + l16;
#pragma unroll
        for (int v = 0; v < 8; ++v) {
          const int qg = q0 + wid * 16 + half * 8 + v;
          if (kvg > qg) s4[nt][v] = -__builtin_inff();
        }
      }
    }

    // online softmax per row (row = fixed v within each 16-lane half)
#pragma unroll
    for (int v = 0; v < 8; ++v) {
      float mx = fmaxf(fmaxf(s4[0][v], s4[1][v]), fmaxf(s4[2][v], s4[3][v]));
      mx = fmaxf(mx, xor16f<1>(mx));
      mx = fmaxf(mx, xor16f<2>(mx));
      mx = fmaxf(mx, xor16f<4>(mx));
      mx = fmaxf(mx, xor16f<8>(mx));
      const float mnew  = fmaxf(mrun[v], mx);
      const float alpha = __expf(mrun[v] - mnew);
      float rs = 0.0f;
#pragma unroll
      for (int nt = 0; nt < 4; ++nt) {
        const float p = __expf(s4[nt][v] - mnew);
        s4[nt][v] = p;
        rs += p;
      }
      rs += xor16f<1>(rs);
      rs += xor16f<2>(rs);
      rs += xor16f<4>(rs);
      rs += xor16f<8>(rs);
      lrun[v] = lrun[v] * alpha + rs;
      mrun[v] = mnew;
#pragma unroll
      for (int nt = 0; nt < 4; ++nt) oacc[nt][v] *= alpha;
    }

    // re-layout P: C-layout registers -> LDS -> A fragments
#pragma unroll
    for (int nt = 0; nt < 4; ++nt)
#pragma unroll
      for (int v = 0; v < 8; ++v)
        Ps[(wid * 16 + half * 8 + v) * LDT + nt * 16 + l16] = (_Float16)s4[nt][v];
    __syncthreads();

    const v16h pf0 = ld_afrag(&Ps[(wid * 16 + l16) * LDT + 0], half);
    const v16h pf1 = ld_afrag(&Ps[(wid * 16 + l16) * LDT + 32], half);
#pragma unroll
    for (int nt = 0; nt < 4; ++nt) {
      const v16h vf0 = ld_bfrag(&Vs[(nt * 16 + l16) * LDT + 0], half);
      const v16h vf1 = ld_bfrag(&Vs[(nt * 16 + l16) * LDT + 32], half);
      oacc[nt] = wmma32(pf0, vf0, oacc[nt]);
      oacc[nt] = wmma32(pf1, vf1, oacc[nt]);
    }
    __syncthreads();  // readers of tile[t] and Ps done -> safe to overwrite
  }

  // normalize, re-layout via LDS (reuse Ps), coalesced f16 stores
#pragma unroll
  for (int v = 0; v < 8; ++v) lrun[v] = 1.0f / lrun[v];
#pragma unroll
  for (int nt = 0; nt < 4; ++nt)
#pragma unroll
    for (int v = 0; v < 8; ++v)
      Ps[(wid * 16 + half * 8 + v) * LDT + nt * 16 + l16] =
          (_Float16)(oacc[nt][v] * lrun[v]);
  __syncthreads();
  {
    uint4* dst = (uint4*)(Ow + (rowbase + q0 + r) * DMODEL + headoff + c0);
    const uint4* src = (const uint4*)(&Ps[r * LDT + c0]);
#pragma unroll
    for (int i = 0; i < 4; ++i) dst[i] = src[i];
  }
}

// ---------------------------------------------------------------------------
extern "C" void kernel_launch(void* const* d_in, const int* in_sizes, int n_in,
                              void* d_out, int out_size, void* d_ws, size_t ws_size,
                              hipStream_t stream) {
  (void)in_sizes; (void)n_in; (void)out_size; (void)ws_size;
  const float* q  = (const float*)d_in[0];
  const float* k  = (const float*)d_in[1];
  const float* v  = (const float*)d_in[2];
  // d_in[3] = causal mask tensor: unused, causality computed analytically
  const float* wq = (const float*)d_in[4];
  const float* wk = (const float*)d_in[5];
  const float* wv = (const float*)d_in[6];
  const float* wo = (const float*)d_in[7];

  const int M = BATCH * SEQ;   // 8192
  const int N = NHEAD * HDIM;  // 1024
  const int K = DMODEL;        // 1024

  const size_t nX = (size_t)M * DMODEL;   // 8388608 elements
  const size_t nW = (size_t)K * N;        // 1048576 elements

  _Float16* Xq  = (_Float16*)d_ws;        // f16 copies of inputs
  _Float16* Xk  = Xq + nX;
  _Float16* Xv  = Xk + nX;
  _Float16* Wtq = Xv + nX;                // f16 transposed weights [N][K]
  _Float16* Wtk = Wtq + nW;
  _Float16* Wtv = Wtk + nW;
  _Float16* Wto = Wtv + nW;
  _Float16* Qw  = Wto + nW;               // Q (pre-scaled), K natural; V transposed
  _Float16* Kw  = Qw + nX;
  _Float16* Vtw = Kw + nX;
  _Float16* Aw  = Vtw + nX;               // attention output (f16)
  // total ws: ~120 MiB of f16

  const int n4 = (int)(nX / 4);
  cvt_f32_f16<<<(n4 + 255) / 256, 256, 0, stream>>>(q, Xq, n4);
  cvt_f32_f16<<<(n4 + 255) / 256, 256, 0, stream>>>(k, Xk, n4);
  cvt_f32_f16<<<(n4 + 255) / 256, 256, 0, stream>>>(v, Xv, n4);

  dim3 tg(K / 64, N / 64);
  transpose_cvt_f16<<<tg, 128, 0, stream>>>(wq, Wtq, K, N);
  transpose_cvt_f16<<<tg, 128, 0, stream>>>(wk, Wtk, K, N);
  transpose_cvt_f16<<<tg, 128, 0, stream>>>(wv, Wtv, K, N);
  transpose_cvt_f16<<<tg, 128, 0, stream>>>(wo, Wto, K, N);

  dim3 gg(M / 128, N / 64);
  gemm_wmma_f16<0><<<gg, 128, 0, stream>>>(Xq, Wtq, Qw, M, N, K, 0.125f);  // 1/sqrt(DK)
  gemm_wmma_f16<0><<<gg, 128, 0, stream>>>(Xk, Wtk, Kw, M, N, K, 1.0f);
  gemm_wmma_f16<2><<<gg, 128, 0, stream>>>(Xv, Wtv, Vtw, M, N, K, 1.0f);

  flash_attn_causal<<<dim3(SEQ / 64, NHEAD, BATCH), 128, 0, stream>>>(Qw, Kw, Vtw, Aw);

  gemm_wmma_f16<1><<<gg, 128, 0, stream>>>(Aw, Wto, d_out, M, N, K, 1.0f);
}
